// AprilCausalAttention_55594056680176
// MI455X (gfx1250) — matrix operand order repaired and checked by reading, not compiled, not run on hardware
//
#include <hip/hip_runtime.h>
#include <hip/hip_bf16.h>

typedef __attribute__((ext_vector_type(16))) _Float16 v16h;
typedef __attribute__((ext_vector_type(8)))  float    v8f;

constexpr int Bn = 4;
constexpr int Tn = 4096;
constexpr int En = 768;
constexpr int Dn = 64;

union Frag {              // 16 halves = one WMMA A/B fragment per lane
    v16h     v;
    _Float16 h[16];
    uint4    u[2];
};
union FAcc {              // 8 floats = one WMMA C/D fragment per lane
    v8f   v;
    float f[8];
};

__device__ __forceinline__ v8f wmma32(v16h a, v16h b, v8f c) {
    // v_wmma_f32_16x16x32_f16: (neg_a, A, neg_b, B, c_mod, C, reuse_a, reuse_b)
    return __builtin_amdgcn_wmma_f32_16x16x32_f16(false, a, false, b, (short)0, c,
                                                  false, false);
}

// ---------------------------------------------------------------------------
// Kernel 0: Wt[w][d][e] = (f16) W_w[e][d]   (3 x 64 x 768)
// ---------------------------------------------------------------------------
__global__ void wtrans_kernel(const float* __restrict__ Wq,
                              const float* __restrict__ Wk,
                              const float* __restrict__ Wv,
                              _Float16* __restrict__ Wt) {
    int idx = blockIdx.x * blockDim.x + threadIdx.x;
    const int total = 3 * Dn * En;
    if (idx >= total) return;
    int w   = idx / (Dn * En);
    int rem = idx - w * (Dn * En);
    int d   = rem / En;
    int e   = rem - d * En;
    const float* W = (w == 0) ? Wq : ((w == 1) ? Wk : Wv);
    Wt[idx] = (_Float16)W[e * Dn + d];
}

// ---------------------------------------------------------------------------
// Kernel 1: q,k,v = x @ W{q,k,v}; q pre-scaled by D^-0.5; v stored transposed.
// One wave per 16-row tile; 24 K-steps x 12 WMMA; B fragments double-buffered
// so each WMMA overlaps the next fragment's global_load_b128 pair.
// ---------------------------------------------------------------------------
__global__ __launch_bounds__(128) void qkv_kernel(
    const float*    __restrict__ x,
    const _Float16* __restrict__ Wt,
    _Float16*       __restrict__ qo,   // [B,T,D] row-major f16 (scaled)
    _Float16*       __restrict__ ko,   // [B,T,D] row-major f16
    _Float16*       __restrict__ vTo)  // [B,D,T] f16
{
    const int lane = threadIdx.x & 31;
    const int lo   = lane & 15;
    const int hi   = lane >> 4;
    const int kb   = hi * 8;               // A-fragment K base
    const int wave = threadIdx.x >> 5;
    const int t0   = (blockIdx.x * 4 + wave) * 16;
    const int b    = blockIdx.y;

    FAcc acc[3][4];
#pragma unroll
    for (int w = 0; w < 3; ++w)
#pragma unroll
        for (int nt = 0; nt < 4; ++nt) {
            v8f z = {};
            acc[w][nt].v = z;
        }

    const float* xrow = x + ((size_t)b * Tn + (t0 + lo)) * En;

    auto loadB = [&](Frag& bf, int f, int k0c) {
        const int w = f >> 2, nt = f & 3;
        const _Float16* wrow =
            Wt + ((size_t)w * Dn + nt * 16 + lo) * En + k0c + hi * 8;
        bf.u[0] = *(const uint4*)(wrow);
        bf.u[1] = *(const uint4*)(wrow + 16);
    };

    Frag bf[2];
    loadB(bf[0], 0, 0);                     // prime the pipeline

    for (int k0 = 0; k0 < En; k0 += 32) {
        // A fragment from x (f32 -> f16), two contiguous 8-float runs per lane
        Frag a;
        const float4 f0 = *(const float4*)(xrow + k0 + kb);
        const float4 f1 = *(const float4*)(xrow + k0 + kb + 4);
        const float4 f2 = *(const float4*)(xrow + k0 + 16 + kb);
        const float4 f3 = *(const float4*)(xrow + k0 + 16 + kb + 4);
        a.h[0]  = (_Float16)f0.x; a.h[1]  = (_Float16)f0.y;
        a.h[2]  = (_Float16)f0.z; a.h[3]  = (_Float16)f0.w;
        a.h[4]  = (_Float16)f1.x; a.h[5]  = (_Float16)f1.y;
        a.h[6]  = (_Float16)f1.z; a.h[7]  = (_Float16)f1.w;
        a.h[8]  = (_Float16)f2.x; a.h[9]  = (_Float16)f2.y;
        a.h[10] = (_Float16)f2.z; a.h[11] = (_Float16)f2.w;
        a.h[12] = (_Float16)f3.x; a.h[13] = (_Float16)f3.y;
        a.h[14] = (_Float16)f3.z; a.h[15] = (_Float16)f3.w;

#pragma unroll
        for (int f = 0; f < 12; ++f) {
            // prefetch next fragment (or fragment 0 of the next K-step)
            if (f < 11)            loadB(bf[(f + 1) & 1], f + 1, k0);
            else if (k0 + 32 < En) loadB(bf[(f + 1) & 1], 0, k0 + 32);
            acc[f >> 2][f & 3].v = wmma32(a.v, bf[f & 1].v, acc[f >> 2][f & 3].v);
        }
    }

#pragma unroll
    for (int nt = 0; nt < 4; ++nt) {
#pragma unroll
        for (int r = 0; r < 8; ++r) {
            const int t = t0 + r + hi * 8;        // C/D row mapping
            const int n = nt * 16 + lo;           // C/D col mapping
            const size_t idx = ((size_t)b * Tn + t) * Dn + n;
            qo[idx] = (_Float16)(acc[0][nt].f[r] * 0.125f);  // fold D^-0.5
            ko[idx] = (_Float16)(acc[1][nt].f[r]);
            vTo[((size_t)b * Dn + n) * Tn + t] = (_Float16)(acc[2][nt].f[r]);
        }
    }
}

// ---------------------------------------------------------------------------
// Kernel 2: register-resident causal flash attention.
// One wave per 16-query tile; key blocks of 32; S computed TRANSPOSED so the
// softmaxed P tile is already in A-fragment layout for the P*V WMMA.
// V-fragment loads are issued before the softmax so they retire behind the
// VALU/trans work; next key block is prefetched (global_prefetch_b8).
// ---------------------------------------------------------------------------
__global__ __launch_bounds__(128) void attn_kernel(
    const _Float16* __restrict__ q,
    const _Float16* __restrict__ k,
    const _Float16* __restrict__ vT,
    float*          __restrict__ out)
{
    const int lane = threadIdx.x & 31;
    const int lo   = lane & 15;
    const int hi   = lane >> 4;
    const int kb   = hi * 8;
    const int wave = threadIdx.x >> 5;
    const int q0   = (blockIdx.x * 4 + wave) * 16;
    const int b    = blockIdx.y;

    // Loop-invariant B fragments of Q^T (lane -> query, K = d in table order)
    Frag qb0, qb1;
    const _Float16* qrow = q + ((size_t)b * Tn + q0 + lo) * Dn;
    qb0.u[0] = *(const uint4*)(qrow + hi * 8);
    qb0.u[1] = *(const uint4*)(qrow + hi * 8 + 16);
    qb1.u[0] = *(const uint4*)(qrow + 32 + hi * 8);
    qb1.u[1] = *(const uint4*)(qrow + 32 + hi * 8 + 16);

    const float NEG = -3.0e38f;
    float Mrow = NEG, Lrow = 0.f;       // per-query stats (dup in lanes g,g+16)
    FAcc o[4];
#pragma unroll
    for (int nt = 0; nt < 4; ++nt) { v8f z = {}; o[nt].v = z; }

    const int qg = q0 + lo;             // query this lane tracks

    for (int nk = 0; nk < q0 + 16; nk += 32) {
        const bool more = (nk + 32 < q0 + 16);

        // ---- batched K A-fragment loads (one clause, one wait) ----
        const _Float16* kr0 = k + ((size_t)b * Tn + nk + lo) * Dn;
        const _Float16* kr1 = kr0 + (size_t)16 * Dn;
        Frag a00, a01, a10, a11;
        a00.u[0] = *(const uint4*)(kr0 + kb);
        a00.u[1] = *(const uint4*)(kr0 + 16 + kb);
        a01.u[0] = *(const uint4*)(kr0 + 32 + kb);
        a01.u[1] = *(const uint4*)(kr0 + 48 + kb);
        a10.u[0] = *(const uint4*)(kr1 + kb);
        a10.u[1] = *(const uint4*)(kr1 + 16 + kb);
        a11.u[0] = *(const uint4*)(kr1 + 32 + kb);
        a11.u[1] = *(const uint4*)(kr1 + 48 + kb);

        // ---- S^T = K_block * Q^T : 2 key tiles x 2 d-steps = 4 WMMA ----
        FAcc st0, st1;
        { v8f z = {}; st0.v = z; st1.v = z; }
        st0.v = wmma32(a00.v, qb0.v, st0.v);
        st0.v = wmma32(a01.v, qb1.v, st0.v);
        st1.v = wmma32(a10.v, qb0.v, st1.v);
        st1.v = wmma32(a11.v, qb1.v, st1.v);

        // ---- issue V B-fragment loads now; they retire during the softmax ---
        Frag vb[4];
#pragma unroll
        for (int nt = 0; nt < 4; ++nt) {
            const _Float16* vr =
                vT + ((size_t)b * Dn + nt * 16 + lo) * Tn + nk + hi * 8;
            vb[nt].u[0] = *(const uint4*)(vr);
            vb[nt].u[1] = *(const uint4*)(vr + 16);
            if (more) __builtin_prefetch(vr + 32, 0, 3);
        }
        if (more) {
            __builtin_prefetch(kr0 + 32 * Dn, 0, 3);   // next key block rows
            __builtin_prefetch(kr1 + 32 * Dn, 0, 3);
        }

        // ---- causal mask + row max (query row = lane pair {lo, lo+16}) ----
        const int kbl = nk + hi * 8;    // key of st*.f[r]
        float m = NEG;
#pragma unroll
        for (int r = 0; r < 8; ++r) {
            if (kbl + r > qg)      st0.f[r] = NEG;
            if (kbl + 16 + r > qg) st1.f[r] = NEG;
            m = fmaxf(m, fmaxf(st0.f[r], st1.f[r]));
        }
        m = fmaxf(m, __shfl_xor(m, 16, 32));
        const float Mnew  = fmaxf(Mrow, m);
        const float alpha = __expf(Mrow - Mnew);

        // ---- P = exp(S - Mnew): already in A-fragment layout ----
        float psum = 0.f;
        Frag p;
#pragma unroll
        for (int r = 0; r < 8; ++r) {
            const float e0 = __expf(st0.f[r] - Mnew);
            const float e1 = __expf(st1.f[r] - Mnew);
            psum += e0 + e1;
            p.h[r]     = (_Float16)e0;   // A K = kb + r      (keys 0..15)
            p.h[8 + r] = (_Float16)e1;   // A K = 16 + kb + r (keys 16..31)
        }
        psum += __shfl_xor(psum, 16, 32);
        Lrow = Lrow * alpha + psum;
        Mrow = Mnew;

        // ---- rescale O rows by alpha (alpha for row m lives in lane m) ----
#pragma unroll
        for (int r = 0; r < 8; ++r) {
            const float ar = __shfl(alpha, r + hi * 8, 32);
            o[0].f[r] *= ar; o[1].f[r] *= ar;
            o[2].f[r] *= ar; o[3].f[r] *= ar;
        }

        // ---- O += P * V : 4 WMMA, operands already resident ----
#pragma unroll
        for (int nt = 0; nt < 4; ++nt)
            o[nt].v = wmma32(p.v, vb[nt].v, o[nt].v);
    }

    // ---- normalize by L and store ----
    const float invL = 1.0f / Lrow;
#pragma unroll
    for (int r = 0; r < 8; ++r) {
        const float inv = __shfl(invL, r + hi * 8, 32);
#pragma unroll
        for (int nt = 0; nt < 4; ++nt) {
            out[((size_t)b * Tn + q0 + r + hi * 8) * Dn + nt * 16 + lo] =
                o[nt].f[r] * inv;
        }
    }
}

// ---------------------------------------------------------------------------
extern "C" void kernel_launch(void* const* d_in, const int* in_sizes, int n_in,
                              void* d_out, int out_size, void* d_ws, size_t ws_size,
                              hipStream_t stream) {
    const float* x  = (const float*)d_in[0];
    const float* Wq = (const float*)d_in[1];
    const float* Wk = (const float*)d_in[2];
    const float* Wv = (const float*)d_in[3];
    float* out = (float*)d_out;

    char* ws = (char*)d_ws;
    constexpr size_t off_wt = 0;                     // 3*64*768*2 = 294,912 B
    constexpr size_t off_q  = 524288;
    constexpr size_t off_k  = off_q + (size_t)Bn * Tn * Dn * 2;
    constexpr size_t off_v  = off_k + (size_t)Bn * Tn * Dn * 2;
    _Float16* Wt  = (_Float16*)(ws + off_wt);
    _Float16* qf  = (_Float16*)(ws + off_q);
    _Float16* kf  = (_Float16*)(ws + off_k);
    _Float16* vTf = (_Float16*)(ws + off_v);

    wtrans_kernel<<<(3 * Dn * En + 255) / 256, 256, 0, stream>>>(Wq, Wk, Wv, Wt);
    qkv_kernel<<<dim3(Tn / 64, Bn), 128, 0, stream>>>(x, Wt, qf, kf, vTf);
    attn_kernel<<<dim3(Tn / 64, Bn), 128, 0, stream>>>(qf, kf, vTf, out);
}
